// GatedDeltaNetBlock_64527588655240
// MI455X (gfx1250) — matrix operand order repaired and checked by reading, not compiled, not run on hardware
//
#include <hip/hip_runtime.h>

// ---------------------------------------------------------------------------
// Gated DeltaNet block forward for MI455X (gfx1250, wave32).
// - All matmuls on v_wmma_f32_16x16x32_bf16 (fp32 accumulate).
// - All LDS tile staging via the Tensor Data Mover (tensor_load_to_lds) with
//   descriptor-driven 2D strided tiles + LDS row padding, completed with
//   s_wait_tensorcnt (TENSORcnt), per CDNA5 ISA ch.7/8 (async_tensor).
// - Weights stored transposed (N x K) so BOTH GEMM fragments read contiguous
//   LDS runs that merge into ds_load_b128.
// ---------------------------------------------------------------------------

typedef __attribute__((ext_vector_type(16))) __bf16 v16bf;
typedef __attribute__((ext_vector_type(8)))  float  v8f;
typedef unsigned int u32;
typedef u32 u32x4 __attribute__((ext_vector_type(4)));
typedef u32 u32x8 __attribute__((ext_vector_type(8)));

#define D_MODEL 1024
#define NHEADS  16
#define HDIM    64
#define CHUNK   64
#define BATCH   4
#define SEQ     4096
#define NCHUNK  (SEQ / CHUNK)          // 64
#define MTOK    (BATCH * SEQ)          // 16384
#define LD      68                     // padded LDS leading dim for 64x64 fp32 tiles
#define TILE_F  (64 * LD)              // floats per padded 64x64 tile

__device__ __forceinline__ v8f wmma_bf16(v16bf a, v16bf b, v8f c) {
  return __builtin_amdgcn_wmma_f32_16x16x32_bf16(false, a, false, b, (short)0, c,
                                                 false, false);
}

// Low 32 bits of a generic pointer to LDS == workgroup LDS byte offset.
__device__ __forceinline__ u32 lds_off(const void* p) {
  return (u32)(unsigned long long)p;
}

// ---------------------------------------------------------------------------
// Tensor Data Mover: load a 2D tile (tile_y rows of tile_x elements, row
// stride row_stride elements in global) into LDS at lds_addr, with per-row
// LDS padding. D# packing per CDNA5 ISA 8.3/8.4:
//   group0: [1:0]=count=1, [63:32]=lds_addr, [120:64]=global_addr, [127:126]=2
//   group1: [17:16]=data_size(log2 bytes), [20]=pad_en, [24:22]=pad_interval
//           (code: 1<<(c+1) DWORDs), [31:25]=pad_amount (DWORDs-1),
//           [79:48]=tensor_dim0, [111:80]=tensor_dim1, [127:112]=tile_dim0,
//           [143:128]=tile_dim1, [207:160]=tensor_dim0_stride
// ---------------------------------------------------------------------------
__device__ __forceinline__ void tdm_load_tile(u32 lds_addr, const void* gsrc,
                                              u32 elem_log2, u32 tile_x, u32 tile_y,
                                              u32 row_stride,
                                              u32 pad_int_code, u32 pad_amt_code) {
  const unsigned long long ga = (unsigned long long)gsrc;
  u32x4 g0;
  g0[0] = 1u;                                              // count=1 (user D#)
  g0[1] = lds_addr;                                        // lds_addr
  g0[2] = (u32)ga;                                         // global_addr[31:0]
  g0[3] = (u32)((ga >> 32) & 0x01FFFFFFu) | (2u << 30);    // addr[56:32] | type=2
  u32x8 g1;
  g1[0] = (elem_log2 << 16) | (1u << 20) | (pad_int_code << 22) | (pad_amt_code << 25);
  g1[1] = (tile_x & 0xFFFFu) << 16;                        // tensor_dim0 = tile_x
  g1[2] = (tile_x >> 16) | ((tile_y & 0xFFFFu) << 16);     // dim0 hi | tensor_dim1 lo
  g1[3] = (tile_y >> 16) | (tile_x << 16);                 // dim1 hi | tile_dim0
  g1[4] = (tile_y & 0xFFFFu);                              // tile_dim1 | tile_dim2=0
  g1[5] = row_stride;                                      // tensor_dim0_stride lo32
  g1[6] = 0u;                                              // stride hi | dim1_stride lo
  g1[7] = 0u;
  asm volatile("tensor_load_to_lds %0, %1" ::"s"(g0), "s"(g1) : "memory");
}

__device__ __forceinline__ void tdm_wait() {
  __builtin_amdgcn_s_wait_tensorcnt((short)0);
}

// ---------------------------------------------------------------------------
// 64x64x64 fp32(LDS) matmul on WMMA, 256 threads (8 wave32), bf16 operands.
// A element (r,k) at A[r*Ars + k*Aks]; B element (k,n) at B[k*Bks + n*Bns].
// C is LD-strided. mode 0: C = AB; 1: C += AB; 2: C = C*cscale + AB.
// CDNA5 wave32 fragment layouts (ISA 7.12.2).
// ---------------------------------------------------------------------------
__device__ __forceinline__ void mm64(const float* A, int Ars, int Aks,
                                     const float* B, int Bks, int Bns,
                                     float* C, int mode, float cscale, int tid) {
  const int lane = tid & 31, wave = tid >> 5;
  const int lr = lane & 15, hi = lane >> 4;
#pragma unroll
  for (int t = 0; t < 2; ++t) {
    const int tile = wave * 2 + t;
    const int tm = (tile >> 2) * 16, tn = (tile & 3) * 16;
    v8f acc = {};
#pragma unroll
    for (int k0 = 0; k0 < 64; k0 += 32) {
      v16bf af, bf;
      const float* as = A + tm * Ars + k0 * Aks;
#pragma unroll
      for (int q = 0; q < 8; ++q) {
        const int kk = (q >> 2) * 16 + hi * 8 + (q & 3) * 2;
        af[2 * q]     = (__bf16)as[lr * Ars + kk * Aks];
        af[2 * q + 1] = (__bf16)as[lr * Ars + (kk + 1) * Aks];
      }
      const float* bs = B + k0 * Bks + tn * Bns;
#pragma unroll
      for (int q = 0; q < 8; ++q) {
        const int kk = (q >> 2) * 16 + hi * 8 + (q & 3) * 2;
        bf[2 * q]     = (__bf16)bs[kk * Bks + lr * Bns];
        bf[2 * q + 1] = (__bf16)bs[(kk + 1) * Bks + lr * Bns];
      }
      acc = wmma_bf16(af, bf, acc);
    }
#pragma unroll
    for (int r = 0; r < 8; ++r) {
      float* cp = &C[(tm + hi * 8 + r) * LD + tn + lr];
      if (mode == 0)       *cp = acc[r];
      else if (mode == 1)  *cp += acc[r];
      else                 *cp = *cp * cscale + acc[r];
    }
  }
}

// ---------------------------------------------------------------------------
// Elementwise / conversion kernels
// ---------------------------------------------------------------------------
__global__ void k_f32_to_bf16(const float* __restrict__ s, __bf16* __restrict__ d,
                              size_t n) {
  size_t i  = (size_t)blockIdx.x * blockDim.x + threadIdx.x;
  size_t st = (size_t)gridDim.x * blockDim.x;
  for (; i < n; i += st) d[i] = (__bf16)s[i];
}

// WcatT (4096 x 1024 bf16): row n = output channel, col k = input channel.
// n in [0,1024) -> Wq col n ; [1024,2048) -> Wk ; etc.
__global__ void k_build_wcat_t(const float* __restrict__ Wq, const float* __restrict__ Wk,
                               const float* __restrict__ Wv, const float* __restrict__ Wg,
                               __bf16* __restrict__ d) {
  int i = blockIdx.x * blockDim.x + threadIdx.x;           // [0, 4096*1024)
  int n = i >> 10, k = i & 1023;
  int sel = n >> 10, col = n & 1023;
  const float* W = (sel == 0) ? Wq : (sel == 1) ? Wk : (sel == 2) ? Wv : Wg;
  d[i] = (__bf16)W[k * D_MODEL + col];
}

// dst (N x K bf16) = transpose of src (K x N fp32)
__global__ void k_transpose_to_bf16(const float* __restrict__ src,
                                    __bf16* __restrict__ dst, int K, int N) {
  int i = blockIdx.x * blockDim.x + threadIdx.x;           // [0, N*K)
  if (i >= N * K) return;
  int n = i / K, k = i - n * K;
  dst[i] = (__bf16)src[(size_t)k * N + n];
}

// ---------------------------------------------------------------------------
// Tiled bf16 WMMA GEMM: Y(MxN fp32) = X(MxKd bf16) @ WT(NxKd bf16)^T
// Block = 256 thr (8 waves), tile 128x128x32; wave computes 32x64 (2x4 tiles).
// Both A and B tiles are (rows x 32) k-major slabs staged by TDM with LDS row
// padding (pad 4 DW every 16 DW -> leading dim 40 bf16), so both fragment
// reads are contiguous 16-byte runs (merge into ds_load_b128).
// ---------------------------------------------------------------------------
__global__ __launch_bounds__(256) void k_gemm_bf16(const __bf16* __restrict__ X,
                                                   const __bf16* __restrict__ WT,
                                                   float* __restrict__ Y,
                                                   int N_, int Kd) {
  __shared__ __align__(16) __bf16 As[128][40];   // 128 rows(m) x 32(k) + 8 pad
  __shared__ __align__(16) __bf16 Bs[128][40];   // 128 rows(n) x 32(k) + 8 pad
  const int tid = threadIdx.x;
  const int lane = tid & 31, wave = tid >> 5;
  const int lr = lane & 15, hi = lane >> 4;
  const int bm = blockIdx.y * 128, bn = blockIdx.x * 128;
  const int wm = (wave & 3) * 32, wn = (wave >> 2) * 64;
  const u32 lds_as = lds_off(&As[0][0]);
  const u32 lds_bs = lds_off(&Bs[0][0]);

  v8f acc[2][4];
#pragma unroll
  for (int i = 0; i < 2; ++i)
#pragma unroll
    for (int j = 0; j < 4; ++j) acc[i][j] = (v8f){};

  for (int k0 = 0; k0 < Kd; k0 += 32) {
    if (wave == 0) {   // one DMA per tile (per-wave instruction, EXEC ignored)
      tdm_load_tile(lds_as, X + (size_t)bm * Kd + k0, 1, 32, 128, (u32)Kd, 3, 3);
      tdm_load_tile(lds_bs, WT + (size_t)bn * Kd + k0, 1, 32, 128, (u32)Kd, 3, 3);
      tdm_wait();
    }
    __syncthreads();

    v16bf af[2], bfr[4];
#pragma unroll
    for (int i = 0; i < 2; ++i)
#pragma unroll
      for (int q = 0; q < 8; ++q) {
        const int kk = (q >> 2) * 16 + hi * 8 + (q & 3) * 2;
        af[i][2 * q]     = As[wm + i * 16 + lr][kk];
        af[i][2 * q + 1] = As[wm + i * 16 + lr][kk + 1];
      }
#pragma unroll
    for (int j = 0; j < 4; ++j)
#pragma unroll
      for (int q = 0; q < 8; ++q) {
        const int kk = (q >> 2) * 16 + hi * 8 + (q & 3) * 2;  // two 8-bf16 runs per lane
        bfr[j][2 * q]     = Bs[wn + j * 16 + lr][kk];
        bfr[j][2 * q + 1] = Bs[wn + j * 16 + lr][kk + 1];
      }
#pragma unroll
    for (int i = 0; i < 2; ++i)
#pragma unroll
      for (int j = 0; j < 4; ++j) acc[i][j] = wmma_bf16(af[i], bfr[j], acc[i][j]);
    __syncthreads();
  }
#pragma unroll
  for (int i = 0; i < 2; ++i)
#pragma unroll
    for (int j = 0; j < 4; ++j)
#pragma unroll
      for (int r = 0; r < 8; ++r)
        Y[(size_t)(bm + wm + i * 16 + hi * 8 + r) * N_ + (bn + wn + j * 16 + lr)] =
            acc[i][j][r];
}

// beta = sigmoid(x@Wb), decay = -exp(A_log)*softplus(x@Wa + dt_bias); (b,h,t) layout
__global__ void k_beta_decay(const float* __restrict__ x, const float* __restrict__ Wb,
                             const float* __restrict__ Wa, const float* __restrict__ dtb,
                             const float* __restrict__ A_log,
                             float* __restrict__ betab, float* __restrict__ decayb) {
  int i = blockIdx.x * blockDim.x + threadIdx.x;
  if (i >= MTOK * NHEADS) return;
  const int h = i & 15, m = i >> 4;
  const float* xr = x + (size_t)m * D_MODEL;
  float db = 0.f, da = 0.f;
  for (int d = 0; d < D_MODEL; ++d) {
    const float xv = xr[d];
    db += xv * Wb[d * NHEADS + h];
    da += xv * Wa[d * NHEADS + h];
  }
  const float beta = 1.f / (1.f + __expf(-db));
  const float z = da + dtb[h];
  const float sp = (z > 20.f) ? z : log1pf(__expf(z));
  const int b = m >> 12, t = m & 4095;
  const size_t o = ((size_t)(b * NHEADS + h)) * SEQ + t;
  betab[o]  = beta;
  decayb[o] = -__expf(A_log[h]) * sp;
}

__global__ void k_cumsum(const float* __restrict__ decayb, float* __restrict__ gbuf) {
  int i = blockIdx.x * blockDim.x + threadIdx.x;
  if (i >= BATCH * NHEADS * NCHUNK) return;
  const size_t base = (size_t)i * CHUNK;
  float s = 0.f;
  for (int c = 0; c < CHUNK; ++c) { s += decayb[base + c]; gbuf[base + c] = s; }
}

__global__ void k_l2norm(float* __restrict__ QKVG) {
  int i = blockIdx.x * blockDim.x + threadIdx.x;
  if (i >= MTOK * NHEADS) return;
  const int h = i & 15, m = i >> 4;
  float* qr = QKVG + (size_t)m * 4096 + h * HDIM;
  float* kr = qr + 1024;
  float sq = 0.f, sk = 0.f;
  for (int d = 0; d < HDIM; ++d) { sq += qr[d] * qr[d]; sk += kr[d] * kr[d]; }
  const float qs = rsqrtf(sq + 1e-6f) * 0.125f;
  const float ks = rsqrtf(sk + 1e-6f);
  for (int d = 0; d < HDIM; ++d) { qr[d] *= qs; kr[d] *= ks; }
}

// ---------------------------------------------------------------------------
// chunk_pre: per (b,h,n) workgroup. Ainv = (I - tril(K_b K^T L))^-1 via
// doubling, then v <- Ainv v, k_cumdecay <- Ainv (k_b e^g).
// k/v tiles (64 rows x 256B, stride 16KB) staged by TDM with pad -> LD=68.
// ---------------------------------------------------------------------------
__global__ __launch_bounds__(256) void k_chunk_pre(const float* __restrict__ QKVG,
                                                   const float* __restrict__ gbuf,
                                                   const float* __restrict__ betab,
                                                   float* __restrict__ vproc,
                                                   float* __restrict__ kcdb) {
  extern __shared__ float sm[];
  __shared__ float sg[CHUNK], sb2[CHUNK];
  const int bid = blockIdx.x;                 // ((b*16+h)*64 + n)
  const int n = bid & 63, h = (bid >> 6) & 15, b = bid >> 10;
  const int tid = threadIdx.x;
  float* sk   = sm;
  float* sv   = sm + 1 * TILE_F;
  float* skb  = sm + 2 * TILE_F;
  float* pk   = sm + 3 * TILE_F;
  float* ainv = sm + 4 * TILE_F;
  float* t1   = sm + 5 * TILE_F;
  float* t2   = sm + 6 * TILE_F;

  const size_t mbase = (size_t)(b * SEQ + n * CHUNK);
  if (tid < 32) {
    const float* ksrc = QKVG + mbase * 4096 + 1024 + (size_t)h * HDIM;
    const float* vsrc = QKVG + mbase * 4096 + 2048 + (size_t)h * HDIM;
    tdm_load_tile(lds_off(sk), ksrc, 2, 64, 64, 4096, 5, 3);
    tdm_load_tile(lds_off(sv), vsrc, 2, 64, 64, 4096, 5, 3);
    tdm_wait();
  }
  if (tid < CHUNK) {
    const size_t gb = ((size_t)(b * NHEADS + h)) * SEQ + n * CHUNK + tid;
    sg[tid]  = gbuf[gb];
    sb2[tid] = betab[gb];
  }
  __syncthreads();
  for (int i = tid; i < 4096; i += 256) {
    const int c = i >> 6, d = i & 63;
    skb[c * LD + d] = sk[c * LD + d] * sb2[c];
    sv[c * LD + d] *= sb2[c];
  }
  __syncthreads();
  mm64(skb, LD, 1, sk, 1, LD, pk, 0, 0.f, tid);       // P = (k b) @ k^T
  __syncthreads();
  for (int i = tid; i < 4096; i += 256) {             // strict tril * L; Ainv = I+P
    const int c = i >> 6, m = i & 63;
    const float p = (c > m) ? pk[c * LD + m] * __expf(sg[c] - sg[m]) : 0.f;
    pk[c * LD + m]   = p;
    ainv[c * LD + m] = p + ((c == m) ? 1.f : 0.f);
  }
  __syncthreads();
  for (int it = 0; it < 5; ++it) {                    // doubling series
    mm64(pk, LD, 1, ainv, LD, 1, t1, 0, 0.f, tid);
    mm64(pk, LD, 1, pk,   LD, 1, t2, 0, 0.f, tid);
    __syncthreads();
    for (int i = tid; i < 4096; i += 256) {
      const int c = i >> 6, d = i & 63;
      ainv[c * LD + d] += t1[c * LD + d];
    }
    { float* tmp = pk; pk = t2; t2 = tmp; }
    __syncthreads();
  }
  for (int i = tid; i < 4096; i += 256) {             // t2 = (k b) e^g
    const int c = i >> 6, d = i & 63;
    t2[c * LD + d] = skb[c * LD + d] * __expf(sg[c]);
  }
  __syncthreads();
  mm64(ainv, LD, 1, sv, LD, 1, t1, 0, 0.f, tid);      // v_out
  mm64(ainv, LD, 1, t2, LD, 1, skb, 0, 0.f, tid);     // k_cumdecay
  __syncthreads();
  {   // vector stores: quarter-row per thread
    const int c = tid >> 2, d0 = (tid & 3) * 16;
    const size_t obase = (size_t)bid * 4096 + c * 64 + d0;
#pragma unroll
    for (int u = 0; u < 4; ++u) {
      *(float4*)&vproc[obase + 4 * u] = *(const float4*)&t1[c * LD + d0 + 4 * u];
      *(float4*)&kcdb[obase + 4 * u]  = *(const float4*)&skb[c * LD + d0 + 4 * u];
    }
  }
}

// ---------------------------------------------------------------------------
// scan: per (b,h) workgroup; sequential over 64 chunks with state S in LDS.
// All four chunk tiles staged per iteration by TDM.
// ---------------------------------------------------------------------------
__global__ __launch_bounds__(256) void k_scan(const float* __restrict__ QKVG,
                                              const float* __restrict__ gbuf,
                                              const float* __restrict__ vproc,
                                              const float* __restrict__ kcdb,
                                              float* __restrict__ obuf) {
  extern __shared__ float sm[];
  __shared__ float sgl[CHUNK];
  const int bh = blockIdx.x;                 // b*16 + h
  const int b = bh >> 4, h = bh & 15;
  const int tid = threadIdx.x;
  float* sS  = sm;
  float* sq  = sm + 1 * TILE_F;
  float* sk2 = sm + 2 * TILE_F;
  float* sv2 = sm + 3 * TILE_F;
  float* skc = sm + 4 * TILE_F;
  float* t1  = sm + 5 * TILE_F;
  float* t2  = sm + 6 * TILE_F;

  for (int i = tid; i < 4096; i += 256) {
    const int c = i >> 6, d = i & 63;
    sS[c * LD + d] = 0.f;
  }
  __syncthreads();

  for (int n = 0; n < NCHUNK; ++n) {
    const size_t mbase = (size_t)(b * SEQ + n * CHUNK);
    const size_t pbase = (((size_t)bh) * NCHUNK + n) * 4096;
    if (tid < 32) {
      tdm_load_tile(lds_off(sq),  QKVG + mbase * 4096 + (size_t)h * HDIM,
                    2, 64, 64, 4096, 5, 3);
      tdm_load_tile(lds_off(sk2), QKVG + mbase * 4096 + 1024 + (size_t)h * HDIM,
                    2, 64, 64, 4096, 5, 3);
      tdm_load_tile(lds_off(sv2), vproc + pbase, 2, 64, 64, 64, 5, 3);
      tdm_load_tile(lds_off(skc), kcdb + pbase,  2, 64, 64, 64, 5, 3);
      tdm_wait();
    }
    if (tid < CHUNK) sgl[tid] = gbuf[((size_t)bh) * SEQ + n * CHUNK + tid];
    __syncthreads();

    mm64(sq, LD, 1, sk2, 1, LD, t1, 0, 0.f, tid);   // raw attn = q @ k^T
    mm64(skc, LD, 1, sS, LD, 1, t2, 0, 0.f, tid);   // kcd @ S
    __syncthreads();
    for (int i = tid; i < 4096; i += 256) {
      const int c = i >> 6, m = i & 63;
      t1[c * LD + m] = (c >= m) ? t1[c * LD + m] * __expf(sgl[c] - sgl[m]) : 0.f;
    }
    for (int i = tid; i < 4096; i += 256) {
      const int c = i >> 6, d = i & 63;
      t2[c * LD + d] = sv2[c * LD + d] - t2[c * LD + d];   // v_new
    }
    __syncthreads();
    for (int i = tid; i < 4096; i += 256) {                 // sv2 <- q * exp(g)
      const int c = i >> 6, d = i & 63;
      sv2[c * LD + d] = sq[c * LD + d] * __expf(sgl[c]);
    }
    __syncthreads();
    mm64(sv2, LD, 1, sS, LD, 1, sq, 0, 0.f, tid);   // o  = (q e^g) @ S
    mm64(t1, LD, 1, t2, LD, 1, sq, 1, 0.f, tid);    // o += attn @ v_new
    __syncthreads();
    {   // vector store of o: quarter-row per thread
      const int c = tid >> 2, d0 = (tid & 3) * 16;
      float* dst = &obuf[(mbase + c) * D_MODEL + h * HDIM + d0];
#pragma unroll
      for (int u = 0; u < 4; ++u)
        *(float4*)&dst[4 * u] = *(const float4*)&sq[c * LD + d0 + 4 * u];
    }
    const float gl = sgl[CHUNK - 1];
    for (int i = tid; i < 4096; i += 256) {                 // sv2 <- k * e^(gl-g)
      const int c = i >> 6, d = i & 63;
      sv2[c * LD + d] = sk2[c * LD + d] * __expf(gl - sgl[c]);
    }
    __syncthreads();
    mm64(sv2, 1, LD, t2, LD, 1, sS, 2, __expf(gl), tid);   // S = S e^gl + k'^T v_new
    __syncthreads();
  }
}

// RMS-norm * norm_w, gate with SiLU, emit bf16 for the output projection
__global__ void k_gate_norm(const float* __restrict__ obuf, const float* __restrict__ QKVG,
                            const float* __restrict__ norm_w, __bf16* __restrict__ ob) {
  int i = blockIdx.x * blockDim.x + threadIdx.x;
  if (i >= MTOK * NHEADS) return;
  const int h = i & 15, m = i >> 4;
  const float* orow = obuf + (size_t)m * D_MODEL + h * HDIM;
  const float* grow = QKVG + (size_t)m * 4096 + 3072 + h * HDIM;
  float s = 0.f;
  for (int d = 0; d < HDIM; ++d) s += orow[d] * orow[d];
  const float sc = rsqrtf(s * (1.f / HDIM) + 1e-5f);
  __bf16* dst = ob + (size_t)m * D_MODEL + h * HDIM;
  for (int d = 0; d < HDIM; ++d) {
    const float g = grow[d];
    const float silu = g / (1.f + __expf(-g));
    dst[d] = (__bf16)(orow[d] * sc * norm_w[d] * silu);
  }
}

// ---------------------------------------------------------------------------
extern "C" void kernel_launch(void* const* d_in, const int* in_sizes, int n_in,
                              void* d_out, int out_size, void* d_ws, size_t ws_size,
                              hipStream_t stream) {
  (void)in_sizes; (void)n_in; (void)out_size; (void)ws_size;
  const float* x    = (const float*)d_in[0];
  const float* Wq   = (const float*)d_in[1];
  const float* Wk   = (const float*)d_in[2];
  const float* Wv   = (const float*)d_in[3];
  const float* Wg   = (const float*)d_in[4];
  const float* Wo   = (const float*)d_in[5];
  const float* Wb   = (const float*)d_in[6];
  const float* Wa   = (const float*)d_in[7];
  const float* dtb  = (const float*)d_in[8];
  const float* Alog = (const float*)d_in[9];
  const float* nw   = (const float*)d_in[10];
  float* out = (float*)d_out;

  char* w = (char*)d_ws;
  auto alloc = [&](size_t bytes) -> void* {
    void* p = (void*)w;
    w += (bytes + 255) & ~(size_t)255;
    return p;
  };
  __bf16* xb    = (__bf16*)alloc((size_t)MTOK * D_MODEL * 2);
  __bf16* wcatT = (__bf16*)alloc((size_t)4096 * D_MODEL * 2);
  __bf16* wobT  = (__bf16*)alloc((size_t)D_MODEL * D_MODEL * 2);
  float*  qkvg  = (float*)alloc((size_t)MTOK * 4096 * 4);
  float*  betab = (float*)alloc((size_t)BATCH * NHEADS * SEQ * 4);
  float*  decb  = (float*)alloc((size_t)BATCH * NHEADS * SEQ * 4);
  float*  gb    = (float*)alloc((size_t)BATCH * NHEADS * SEQ * 4);
  float*  vproc = (float*)alloc((size_t)BATCH * NHEADS * SEQ * HDIM * 4);
  float*  kcdb  = (float*)alloc((size_t)BATCH * NHEADS * SEQ * HDIM * 4);
  float*  obuf  = (float*)alloc((size_t)MTOK * D_MODEL * 4);
  __bf16* ob    = (__bf16*)alloc((size_t)MTOK * D_MODEL * 2);

  const size_t smem = (size_t)7 * TILE_F * sizeof(float);

  k_f32_to_bf16<<<16384, 256, 0, stream>>>(x, xb, (size_t)MTOK * D_MODEL);
  k_build_wcat_t<<<(4096 * D_MODEL) / 256, 256, 0, stream>>>(Wq, Wk, Wv, Wg, wcatT);
  k_transpose_to_bf16<<<(D_MODEL * D_MODEL) / 256, 256, 0, stream>>>(Wo, wobT,
                                                                     D_MODEL, D_MODEL);

  k_gemm_bf16<<<dim3(4096 / 128, MTOK / 128), 256, 0, stream>>>(xb, wcatT, qkvg, 4096,
                                                                D_MODEL);

  k_beta_decay<<<(MTOK * NHEADS) / 256, 256, 0, stream>>>(x, Wb, Wa, dtb, Alog, betab,
                                                          decb);
  k_cumsum<<<(BATCH * NHEADS * NCHUNK) / 256, 256, 0, stream>>>(decb, gb);
  k_l2norm<<<(MTOK * NHEADS) / 256, 256, 0, stream>>>(qkvg);

  k_chunk_pre<<<BATCH * NHEADS * NCHUNK, 256, smem, stream>>>(qkvg, gb, betab, vproc,
                                                              kcdb);
  k_scan<<<BATCH * NHEADS, 256, smem, stream>>>(qkvg, gb, vproc, kcdb, obuf);

  k_gate_norm<<<(MTOK * NHEADS) / 256, 256, 0, stream>>>(obuf, qkvg, nw, ob);
  k_gemm_bf16<<<dim3(D_MODEL / 128, MTOK / 128), 256, 0, stream>>>(ob, wobT, out,
                                                                   D_MODEL, D_MODEL);
}
